// PaiNN_31825707663885
// MI455X (gfx1250) — compile-verified, matrix operand order checked by reading, not run on hardware
//
#include <hip/hip_runtime.h>
#include <hip/hip_bf16.h>

typedef __attribute__((ext_vector_type(16))) __bf16 v16bf;
typedef __attribute__((ext_vector_type(8)))  float  v8f;

#define KF    128
#define NRBF  20
#define WCOLS 1152      // L * 3 * K
#define CUT_R 5.0f

union FragB { v16bf v; unsigned int u[8]; };
union AccU  { v8f   v; float        f[8]; };

__device__ __forceinline__ unsigned short f2bf(float f) {
    unsigned int u = __float_as_uint(f);
    u += 0x7fffu + ((u >> 16) & 1u);   // round-to-nearest-even
    return (unsigned short)(u >> 16);
}

// ---------------- weight transpose + convert: W (Kin,Nout) f32 -> WT (Nout,Kin) bf16
__global__ void wconv_t(const float* __restrict__ W, unsigned short* __restrict__ WT,
                        int Kin, int Nout) {
    int i = blockIdx.x * blockDim.x + threadIdx.x;
    int total = Kin * Nout;
    if (i >= total) return;
    int k = i / Nout, n = i - k * Nout;
    WT[(size_t)n * Kin + k] = f2bf(W[i]);
}

__global__ void zero_f32(float* __restrict__ p, long long n) {
    long long i = (long long)blockIdx.x * blockDim.x + threadIdx.x;
    long long s = (long long)gridDim.x * blockDim.x;
    for (; i < n; i += s) p[i] = 0.0f;
}

// ---------------- q = emb[x] (f32 + bf16 shadow), mu = 0
__global__ void embed_k(const int* __restrict__ x, const float* __restrict__ emb,
                        float* __restrict__ q, unsigned short* __restrict__ qb,
                        float* __restrict__ mu) {
    int n = blockIdx.x, ch = threadIdx.x;
    float v = emb[(size_t)x[n] * KF + ch];
    q [(size_t)n * KF + ch] = v;
    qb[(size_t)n * KF + ch] = f2bf(v);
#pragma unroll
    for (int c = 0; c < 3; ++c) mu[((size_t)n * 3 + c) * KF + ch] = 0.0f;
}

// ---------------- per-edge geometry: dir, phi*fcut, fcut
__global__ void edge_geom(const int* __restrict__ ei, const float* __restrict__ pos,
                          float* __restrict__ dir, float* __restrict__ prbf,
                          float* __restrict__ fct, int E) {
    int e = blockIdx.x * blockDim.x + threadIdx.x;
    if (e >= E) return;
    int i = ei[e], j = ei[E + e];
    float rx = pos[i * 3 + 0] - pos[j * 3 + 0];
    float ry = pos[i * 3 + 1] - pos[j * 3 + 1];
    float rz = pos[i * 3 + 2] - pos[j * 3 + 2];
    float d  = sqrtf(rx * rx + ry * ry + rz * rz);
    float inv = 1.0f / d;
    dir[e * 3 + 0] = rx * inv;
    dir[e * 3 + 1] = ry * inv;
    dir[e * 3 + 2] = rz * inv;
    float fc = (d < CUT_R) ? 0.5f * (__cosf(d * 3.14159265358979f / CUT_R) + 1.0f) : 0.0f;
    fct[e] = fc;
    const float spacing = CUT_R / (float)(NRBF - 1);
    const float coeff   = -0.5f / (spacing * spacing);
#pragma unroll
    for (int k = 0; k < NRBF; ++k) {
        float t = d - spacing * (float)k;
        prbf[(size_t)e * NRBF + k] = __expf(coeff * t * t) * fc;
    }
}

// ---------------- WMMA GEMM: out(M,Nout) = act( A(M,Kin)bf16 @ W(Kin,Nout) + bias )
// A is bf16 (M,Kin); staged into LDS with async-to-LDS DMA. WT is bf16 (Nout,Kin).
// 4 waves/block, each wave owns a 16x16 tile, K stepped by 32.
__global__ void gemm_bf16_wmma(const unsigned short* __restrict__ Ab,
                               const unsigned short* __restrict__ WT,
                               const float* __restrict__ bias,
                               float* __restrict__ outF, unsigned short* __restrict__ outB,
                               int M, int Kin, int Nout, int act) {
    __shared__ unsigned short sA[16 * (256 + 8)];
    const int sStride = Kin + 8;                 // +16B pad: keeps 16B chunks aligned, rotates banks
    const int tid   = threadIdx.x;
    const int mBase = blockIdx.x * 16;
    const int wave  = tid >> 5;
    const int lane  = tid & 31;
    const int nLoc  = lane & 15;
    const int hi    = (lane >> 4) & 1;
    const int tileN = blockIdx.y * 64 + wave * 16;

    // ---- async DMA stage of the 16 x Kin bf16 A tile into LDS (16B chunks)
    const int cpr = Kin >> 3;                    // 16B chunks per row
    const unsigned int ldsBase = (unsigned int)(unsigned long long)(const void*)sA;
    for (int c = tid; c < 16 * cpr; c += blockDim.x) {
        int row = c / cpr, col = c - row * cpr;
        int m = mBase + row; if (m >= M) m = M - 1;
        unsigned long long g =
            (unsigned long long)(const void*)(Ab + (size_t)m * Kin + (col << 3));
        unsigned int l = ldsBase + (unsigned int)((row * sStride + (col << 3)) * 2);
        asm volatile("global_load_async_to_lds_b128 %0, %1, off"
                     :: "v"(l), "v"(g) : "memory");
    }
    asm volatile("s_wait_asynccnt 0x0" ::: "memory");
    __syncthreads();

    AccU acc;
#pragma unroll
    for (int r = 0; r < 8; ++r) acc.f[r] = 0.0f;

    const unsigned short* aPtr = &sA[(lane & 15) * sStride];
    const unsigned short* bPtr = WT + (size_t)(tileN + nLoc) * Kin;

    for (int kb = 0; kb < Kin; kb += 32) {
        FragB fa, fb;
#pragma unroll
        for (int j = 0; j < 8; ++j) {
            // ISA 16-bit A/B layout: lanes 0-15 K=0..7,16..23 ; lanes 16-31 K=8..15,24..31
            int k = kb + ((j & 4) << 2) + (hi << 3) + ((j & 3) << 1);
            fa.u[j] = *(const unsigned int*)(aPtr + k);   // ds load (LDS)
            fb.u[j] = *(const unsigned int*)(bPtr + k);   // global load
        }
        acc.v = __builtin_amdgcn_wmma_f32_16x16x32_bf16(
                    false, fa.v, false, fb.v, (short)0, acc.v, false, false);
    }

    const int n = tileN + nLoc;
#pragma unroll
    for (int r = 0; r < 8; ++r) {
        int m = mBase + r + hi * 8;
        if (m >= M) continue;
        float v = acc.f[r];
        if (bias) v += bias[n];
        if (act)  v = v / (1.0f + __expf(-v));           // silu
        if (outF) outF[(size_t)m * Nout + n] = v;
        if (outB) outB[(size_t)m * Nout + n] = f2bf(v);
    }
}

// ---------------- per-edge message + scatter (1 block / edge, 128 channels)
__global__ void edge_msg(int l, const int* __restrict__ ei,
                         const float* __restrict__ prbf, const float* __restrict__ fct,
                         const float* __restrict__ dir,
                         const float* __restrict__ filt_w, const float* __restrict__ filt_b,
                         const float* __restrict__ xfeat, const float* __restrict__ mu,
                         float* __restrict__ dq, float* __restrict__ dmu, int E) {
    __shared__ float sp[NRBF];
    __shared__ float sd[3];
    __shared__ float sfc;
    __shared__ int   sij[2];
    const int e  = blockIdx.x;
    const int ch = threadIdx.x;
    if      (ch < NRBF)      sp[ch] = prbf[(size_t)e * NRBF + ch];
    else if (ch < NRBF + 3)  sd[ch - NRBF] = dir[e * 3 + (ch - NRBF)];
    else if (ch == NRBF + 3) sfc = fct[e];
    else if (ch == NRBF + 4) sij[0] = ei[e];
    else if (ch == NRBF + 5) sij[1] = ei[E + e];
    __syncthreads();

    const int i = sij[0], j = sij[1];
    const float* xj  = xfeat + (size_t)j * 3 * KF;
    const float* muj = mu    + (size_t)j * 3 * KF;
    // warm the scattered gather lines while the filter dot-product runs
    __builtin_prefetch(&xj[ch], 0, 1);
    __builtin_prefetch(&xj[2 * KF + ch], 0, 1);
    __builtin_prefetch(&muj[ch], 0, 1);
    __builtin_prefetch(&muj[2 * KF + ch], 0, 1);

    const int cBase = l * 3 * KF + ch;
    float w0 = filt_b[cBase         ] * sfc;
    float w1 = filt_b[cBase +     KF] * sfc;
    float w2 = filt_b[cBase + 2 * KF] * sfc;
    const float* fw = filt_w + cBase;
#pragma unroll
    for (int k = 0; k < NRBF; ++k) {
        float p = sp[k];
        const float* row = fw + (size_t)k * WCOLS;
        w0 += p * row[0];
        w1 += p * row[KF];
        w2 += p * row[2 * KF];
    }
    float mq = xj[ch]          * w0;     // dq_e
    float mR = xj[KF + ch]     * w1;     // dmuR
    float mM = xj[2 * KF + ch] * w2;     // dmumu
    atomicAdd(&dq[(size_t)i * KF + ch], mq);
#pragma unroll
    for (int c = 0; c < 3; ++c)
        atomicAdd(&dmu[((size_t)i * 3 + c) * KF + ch],
                  mR * sd[c] + mM * muj[c * KF + ch]);
}

// ---------------- q += dq ; mu += dmu (and refresh bf16 shadow of mu)
__global__ void apply_acc(float* __restrict__ q, const float* __restrict__ dq,
                          float* __restrict__ mu, const float* __restrict__ dmu,
                          unsigned short* __restrict__ mub) {
    int n = blockIdx.x, ch = threadIdx.x;
    q[(size_t)n * KF + ch] += dq[(size_t)n * KF + ch];
#pragma unroll
    for (int c = 0; c < 3; ++c) {
        size_t idx = ((size_t)n * 3 + c) * KF + ch;
        float v = mu[idx] + dmu[idx];
        mu[idx]  = v;
        mub[idx] = f2bf(v);
    }
}

// ---------------- ctx = [q, ||mu_V||] (bf16, feeds next GEMM) from mu_mix (N,3,256)
__global__ void node_ctx(const float* __restrict__ q, const float* __restrict__ mu_mix,
                         unsigned short* __restrict__ ctxb) {
    int n = blockIdx.x, ch = threadIdx.x;
    const float* mm = mu_mix + (size_t)n * 3 * 256;
    float s = 1e-8f;
#pragma unroll
    for (int c = 0; c < 3; ++c) { float v = mm[c * 256 + ch]; s += v * v; }
    ctxb[(size_t)n * 256 + ch]      = f2bf(q[(size_t)n * KF + ch]);
    ctxb[(size_t)n * 256 + KF + ch] = f2bf(sqrtf(s));
}

// ---------------- final per-node update (refreshes q's bf16 shadow)
__global__ void node_update(const float* __restrict__ mu_mix, const float* __restrict__ x2,
                            float* __restrict__ q, unsigned short* __restrict__ qb,
                            float* __restrict__ mu) {
    int n = blockIdx.x, ch = threadIdx.x;
    const float* mm = mu_mix + (size_t)n * 3 * 256;
    float dot = 0.0f, muW[3];
#pragma unroll
    for (int c = 0; c < 3; ++c) {
        float v = mm[c * 256 + ch];
        float w = mm[c * 256 + KF + ch];
        muW[c] = w;
        dot += v * w;
    }
    const float* xr = x2 + (size_t)n * 3 * KF;
    float dq_i = xr[ch], dmu_i = xr[KF + ch], dqmu_i = xr[2 * KF + ch];
    float qn = q[(size_t)n * KF + ch] + dq_i + dqmu_i * dot;
    q [(size_t)n * KF + ch] = qn;
    qb[(size_t)n * KF + ch] = f2bf(qn);
#pragma unroll
    for (int c = 0; c < 3; ++c)
        mu[((size_t)n * 3 + c) * KF + ch] += dmu_i * muW[c];
}

// ---------------- graph pooling
__global__ void pool_k(const int* __restrict__ batch, const float* __restrict__ q,
                       float* __restrict__ out) {
    int n = blockIdx.x, ch = threadIdx.x;
    atomicAdd(&out[(size_t)batch[n] * KF + ch], q[(size_t)n * KF + ch]);
}

extern "C" void kernel_launch(void* const* d_in, const int* in_sizes, int n_in,
                              void* d_out, int out_size, void* d_ws, size_t ws_size,
                              hipStream_t stream) {
    const int*   x      = (const int*)  d_in[0];
    const float* pos    = (const float*)d_in[1];
    const int*   eidx   = (const int*)  d_in[2];
    const int*   batch  = (const int*)  d_in[3];
    const float* emb    = (const float*)d_in[4];
    const float* filt_w = (const float*)d_in[5];
    const float* filt_b = (const float*)d_in[6];
    const float* iw1    = (const float*)d_in[7];
    const float* ib1    = (const float*)d_in[8];
    const float* iw2    = (const float*)d_in[9];
    const float* ib2    = (const float*)d_in[10];
    const float* muw    = (const float*)d_in[11];
    const float* mw1    = (const float*)d_in[12];
    const float* mb1    = (const float*)d_in[13];
    const float* mw2    = (const float*)d_in[14];
    const float* mb2    = (const float*)d_in[15];

    const int N = in_sizes[0];
    const int E = in_sizes[2] / 2;
    const long long NK = (long long)N * 128;

    // ---- workspace carve-up: f32 region, then bf16 region (16B-aligned)
    float* ws   = (float*)d_ws;
    float* q      = ws; ws += NK;
    float* mu     = ws; ws += 3 * NK;
    float* dq     = ws; ws += NK;
    float* dmu    = ws; ws += 3 * NK;
    float* xfeat  = ws; ws += 3 * NK;
    float* mu_mix = ws; ws += 6 * NK;     // (N,3,256)
    float* x2     = ws; ws += 3 * NK;
    float* dir    = ws; ws += 3LL * E;
    float* prbf   = ws; ws += (long long)NRBF * E;
    float* fct    = ws; ws += E;
    unsigned short* qb   = (unsigned short*)ws;
    unsigned short* mub  = qb   + NK;
    unsigned short* hb   = mub  + 3 * NK;
    unsigned short* ctxb = hb   + NK;
    unsigned short* h2b  = ctxb + 2 * NK;
    unsigned short* w1t  = h2b  + NK;
    unsigned short* w2t  = w1t  + 3LL * 128 * 128;
    unsigned short* muwT = w2t  + 3LL * 384 * 128;
    unsigned short* mw1T = muwT + 3LL * 256 * 128;
    unsigned short* mw2T = mw1T + 3LL * 128 * 256;

    // weight transpose + bf16 convert (tiny; done every call for determinism)
    for (int l = 0; l < 3; ++l) {
        wconv_t<<<(128 * 128 + 255) / 256, 256, 0, stream>>>(iw1 + (size_t)l * 128 * 128, w1t  + (size_t)l * 128 * 128, 128, 128);
        wconv_t<<<(128 * 384 + 255) / 256, 256, 0, stream>>>(iw2 + (size_t)l * 128 * 384, w2t  + (size_t)l * 384 * 128, 128, 384);
        wconv_t<<<(128 * 256 + 255) / 256, 256, 0, stream>>>(muw + (size_t)l * 128 * 256, muwT + (size_t)l * 256 * 128, 128, 256);
        wconv_t<<<(256 * 128 + 255) / 256, 256, 0, stream>>>(mw1 + (size_t)l * 256 * 128, mw1T + (size_t)l * 128 * 256, 256, 128);
        wconv_t<<<(128 * 384 + 255) / 256, 256, 0, stream>>>(mw2 + (size_t)l * 128 * 384, mw2T + (size_t)l * 384 * 128, 128, 384);
    }

    edge_geom<<<(E + 255) / 256, 256, 0, stream>>>(eidx, pos, dir, prbf, fct, E);
    embed_k<<<N, 128, 0, stream>>>(x, emb, q, qb, mu);

    const int gM  = (N + 15) / 16;
    const int gM3 = (3 * N + 15) / 16;

    for (int l = 0; l < 3; ++l) {
        // h(bf16) = silu(q @ W1 + b1)
        gemm_bf16_wmma<<<dim3(gM, 2), 128, 0, stream>>>(qb, w1t + (size_t)l * 128 * 128,
                                                        ib1 + l * 128, nullptr, hb,
                                                        N, 128, 128, 1);
        // xfeat(f32) = h @ W2 + b2
        gemm_bf16_wmma<<<dim3(gM, 6), 128, 0, stream>>>(hb, w2t + (size_t)l * 384 * 128,
                                                        ib2 + l * 384, xfeat, nullptr,
                                                        N, 128, 384, 0);
        zero_f32<<<1024, 256, 0, stream>>>(dq, NK);
        zero_f32<<<2048, 256, 0, stream>>>(dmu, 3 * NK);
        edge_msg<<<E, 128, 0, stream>>>(l, eidx, prbf, fct, dir, filt_w, filt_b,
                                        xfeat, mu, dq, dmu, E);
        apply_acc<<<N, 128, 0, stream>>>(q, dq, mu, dmu, mub);
        // mu_mix(f32, (3N,256)) = mu @ mix_muw (no bias)
        gemm_bf16_wmma<<<dim3(gM3, 4), 128, 0, stream>>>(mub, muwT + (size_t)l * 256 * 128,
                                                         nullptr, mu_mix, nullptr,
                                                         3 * N, 128, 256, 0);
        node_ctx<<<N, 128, 0, stream>>>(q, mu_mix, ctxb);
        // h2(bf16) = silu(ctx @ mix_w1 + b1)
        gemm_bf16_wmma<<<dim3(gM, 2), 128, 0, stream>>>(ctxb, mw1T + (size_t)l * 128 * 256,
                                                        mb1 + l * 128, nullptr, h2b,
                                                        N, 256, 128, 1);
        // x2(f32) = h2 @ mix_w2 + b2
        gemm_bf16_wmma<<<dim3(gM, 6), 128, 0, stream>>>(h2b, mw2T + (size_t)l * 384 * 128,
                                                        mb2 + l * 384, x2, nullptr,
                                                        N, 128, 384, 0);
        node_update<<<N, 128, 0, stream>>>(mu_mix, x2, q, qb, mu);
    }

    zero_f32<<<16, 256, 0, stream>>>((float*)d_out, (long long)out_size);
    pool_k<<<N, 128, 0, stream>>>(batch, q, (float*)d_out);
}
